// AttentionCPCEncoder_38585986187650
// MI455X (gfx1250) — matrix operand order repaired and checked by reading, not compiled, not run on hardware
//
#include <hip/hip_runtime.h>

// ---------------------------------------------------------------------------
// CDNA5 (gfx1250) bf16 WMMA implementation of AttentionCPCEncoder forward.
// Every dense op runs through v_wmma_f32_16x16x32_bf16 (wave32).
// Each wave computes a 16x64 output strip (4 accumulators, shared A fragment).
// All waves of a block share the same 16-row A tile: wave 0 stages it into
// LDS with global_load_async_to_lds_b128 (ASYNCcnt), everyone reads it back
// with ds_load_b128 -> up to 8x less A traffic. B^T fragments load as packed
// dwords (global_load_b128 quads).
// ---------------------------------------------------------------------------

typedef __attribute__((ext_vector_type(16))) __bf16 v16bf;
typedef __attribute__((ext_vector_type(8)))  float  v8f;

#define DEV __device__ __forceinline__

DEV unsigned short f2b(float f) {
  union { float f; unsigned int i; } u; u.f = f;
  unsigned int r = (u.i + 0x7FFFu + ((u.i >> 16) & 1u)) >> 16;  // RNE
  return (unsigned short)r;
}
DEV float gelu_f(float x) {  // tanh approximation (flax approximate=True)
  float x3 = x * x * x;
  return 0.5f * x * (1.0f + tanhf(0.7978845608028654f * (x + 0.044715f * x3)));
}
DEV float sigm(float x) { return 1.0f / (1.0f + expf(-x)); }

enum {
  M_PLAIN = 0, M_BIAS = 1, M_BN_GELU = 2, M_ACCUM = 3, M_ALPHA = 4,
  M_BIAS_BF16 = 5, M_BF16 = 6, M_BIAS_RES = 7, M_BIAS_GELU_BF16 = 8,
  M_BIAS_BF16_VT = 9
};

union AFrag { v16bf v; unsigned int u[8]; };

// ---------------------------------------------------------------------------
// bf16 WMMA GEMM: C[M,N] = epilogue(A[M,K] * B[K,N]).
// B is ALWAYS stored transposed [N,K].
// ROWMAP: logical row r reads physical row (r/512)*rowOuter + (r%512).
// blockIdx.y batches with element strides bsA/bsB/bsC (attention heads).
// Launch contract: blockDim.x = wpb*32 where wpb divides N/64; grid.x =
// (M/16)*((N/64)/wpb). All waves in a block share the A row-tile tm.
// ---------------------------------------------------------------------------
template<int MODE, bool ROWMAP>
__global__ void ae_gemm(const unsigned short* __restrict__ A,
                        const unsigned short* __restrict__ B,
                        float* __restrict__ outF,
                        unsigned short* __restrict__ outB,
                        const float* __restrict__ bias,
                        const float* __restrict__ scale,
                        const float* __restrict__ res,
                        int M, int N, int K,
                        int lda, int ldb, int ldc,
                        float alpha, int rowOuter,
                        long long bsA, long long bsB, long long bsC)
{
  __shared__ unsigned short atile[512];          // 16x32 bf16 A tile, fragment order

  const int lane = threadIdx.x & 31;
  const int w    = threadIdx.x >> 5;
  const int wpb  = blockDim.x >> 5;
  const int nt4  = N >> 6;                       // 64-wide strips
  const int ngrp = nt4 / wpb;
  const int tm   = blockIdx.x / ngrp;
  const int tn   = (blockIdx.x % ngrp) * wpb + w;
  const int half = lane >> 4, l16 = lane & 15;

  const unsigned short* Ab = A + (size_t)blockIdx.y * bsA;
  const unsigned short* Bb = B + (size_t)blockIdx.y * bsB;
  const size_t cOff = (size_t)blockIdx.y * bsC;

  int arow = tm * 16 + l16;
  if (ROWMAP) arow = (arow >> 9) * rowOuter + (arow & 511);
  // Lane's A fragment = two contiguous 16B runs: [arow, k0+8h .. +7] and +16.
  const unsigned short* gA = Ab + (size_t)arow * lda + (half << 3);
  // B fragment rows: physical row (tn*64 + i*16 + l16) of B^T, ld = ldb elems.
  const unsigned int* bp = (const unsigned int*)(Bb + (size_t)(tn * 64 + l16) * ldb);
  const int bstep = (ldb << 3);                  // 16 rows in dwords

  // LDS byte address of this lane's 32B fragment slot (addr[31:0] == LDS off).
  const unsigned myl = (unsigned)(unsigned long long)(const void*)atile + (lane << 5);

  v8f acc[4] = {};
  for (int k0 = 0; k0 < K; k0 += 32) {
    // wave 0: kick async copy of the shared 16x32 A tile into LDS
    if (threadIdx.x < 32) {
      const unsigned long long g0 = (unsigned long long)(gA + k0);
      asm volatile("global_load_async_to_lds_b128 %0, %1, off"
                   :: "v"(myl), "v"(g0) : "memory");
      asm volatile("global_load_async_to_lds_b128 %0, %1, off"
                   :: "v"(myl + 16u), "v"(g0 + 32ull) : "memory");
    }
    // B fragments from global (overlaps the async copy)
    AFrag b[4];
#pragma unroll
    for (int i = 0; i < 4; ++i) {
#pragma unroll
      for (int j = 0; j < 8; ++j) {
        const int kb = k0 + (half << 3) + (j << 1) + ((j >> 2) << 3);   // 32x16: VGPR j -> K=8h+2j (+8 for j>=4)
        b[i].u[j] = bp[i * bstep + (kb >> 1)];
      }
    }
    if (k0 + 32 < K)
      __builtin_prefetch((const void*)(bp + ((k0 + 32) >> 1)), 0, 0);
    if (threadIdx.x < 32)
      asm volatile("s_wait_asynccnt 0" ::: "memory");
    __syncthreads();
    // every wave reads its own fragment slot back (2x ds_load_b128)
    AFrag a;
    const uint4* lp = (const uint4*)atile + (lane << 1);
    uint4 c0 = lp[0], c1 = lp[1];
    a.u[0] = c0.x; a.u[1] = c0.y; a.u[2] = c0.z; a.u[3] = c0.w;
    a.u[4] = c1.x; a.u[5] = c1.y; a.u[6] = c1.z; a.u[7] = c1.w;
#pragma unroll
    for (int i = 0; i < 4; ++i) {
      acc[i] = __builtin_amdgcn_wmma_f32_16x16x32_bf16(
          /*neg_a=*/false, a.v, /*neg_b=*/false, b[i].v,
          /*c_mod=*/(short)0, acc[i], /*reuse_a=*/false, /*reuse_b=*/false);
    }
    __syncthreads();
  }

  // C/D 16x16 f32 layout: VGPR r -> row 8h+r, column = lane&15.
#pragma unroll
  for (int i = 0; i < 4; ++i) {
#pragma unroll
    for (int r = 0; r < 8; ++r) {
      const int row = tm * 16 + (half << 3) + r;
      const int col = tn * 64 + i * 16 + l16;
      const size_t idx = cOff + (size_t)row * ldc + col;
      const float v = acc[i][r];
      if constexpr (MODE == M_PLAIN) {
        outF[idx] = v;
      } else if constexpr (MODE == M_BIAS) {
        outF[idx] = v + bias[col];
      } else if constexpr (MODE == M_BN_GELU) {
        const float g = gelu_f(v * scale[col] + bias[col]);
        if (outB) outB[idx] = f2b(g);
        if (outF) outF[idx] = g;
      } else if constexpr (MODE == M_ACCUM) {
        outF[idx] += alpha * (v + bias[col]);
      } else if constexpr (MODE == M_ALPHA) {
        outF[idx] = alpha * v;
      } else if constexpr (MODE == M_BIAS_BF16) {
        outB[idx] = f2b(v + bias[col]);
      } else if constexpr (MODE == M_BF16) {
        outB[idx] = f2b(v);
      } else if constexpr (MODE == M_BIAS_RES) {
        outF[idx] = v + bias[col] + res[idx];
      } else if constexpr (MODE == M_BIAS_GELU_BF16) {
        outB[idx] = f2b(gelu_f(v + bias[col]));
      } else if constexpr (MODE == M_BIAS_BF16_VT) {
        // V projection: write V transposed per head, vT[b, h, d, s]
        // (row = b*512 + s over 16384; col = h*128 + d over 1024).
        const size_t tidx =
            ((size_t)((row >> 9) * 8 + (col >> 7)) * 128 + (col & 127)) * 512 +
            (row & 511);
        outB[tidx] = f2b(v + bias[col]);
      }
    }
  }
}

// ---------------------------------------------------------------------------
// Elementwise / prep kernels
// ---------------------------------------------------------------------------
__global__ void ae_prep_input(const float* __restrict__ x, unsigned short* __restrict__ o, int n) {
  int i = blockIdx.x * blockDim.x + threadIdx.x;
  if (i < n) o[i] = f2b(x[i] * 1e20f);
}
__global__ void ae_cast(const float* __restrict__ in, unsigned short* __restrict__ o, int n) {
  int i = blockIdx.x * blockDim.x + threadIdx.x;
  if (i < n) o[i] = f2b(in[i]);
}
__global__ void ae_zero_bf(unsigned short* __restrict__ o, int n) {
  int i = blockIdx.x * blockDim.x + threadIdx.x;
  if (i < n) o[i] = 0;
}
// out[n*ldout + k] = bf16(in[k*N + n]); weights stored transposed for GEMM.
__global__ void ae_cast_t(const float* __restrict__ in, unsigned short* __restrict__ o,
                          int K, int N, int ldout) {
  int i = blockIdx.x * blockDim.x + threadIdx.x;
  if (i >= K * N) return;
  int k = i / N, n = i % N;
  o[(size_t)n * ldout + k] = f2b(in[i]);
}
// Fold conv bias + BN into per-channel scale/shift.
__global__ void ae_bnprep(const float* __restrict__ cb, const float* __restrict__ s,
                          const float* __restrict__ bb, const float* __restrict__ m,
                          const float* __restrict__ v, float* __restrict__ scale,
                          float* __restrict__ shift, int C) {
  int c = blockIdx.x * blockDim.x + threadIdx.x;
  if (c >= C) return;
  float sc = s[c] * rsqrtf(v[c] + 1e-5f);
  scale[c] = sc;
  shift[c] = bb[c] + (cb[c] - m[c]) * sc;
}
// im2col for conv1d k=9, stride 2, SAME (pad_lo=3): col[(b,t), u*Cin+c] = in[b, 2t+u-3, c]
__global__ void ae_im2col(const unsigned short* __restrict__ in, unsigned short* __restrict__ o,
                          int Bn, int Lin, int Cin, int Lout, int Kpad) {
  int i = blockIdx.x * blockDim.x + threadIdx.x;
  int total = Bn * Lout * Kpad;
  if (i >= total) return;
  int kk = i % Kpad;
  int r  = i / Kpad;
  int t  = r % Lout;
  int b  = r / Lout;
  int u = kk / Cin, c = kk % Cin;
  int pos = 2 * t + u - 3;
  unsigned short val = 0;
  if (u < 9 && pos >= 0 && pos < Lin) val = in[((size_t)b * Lin + pos) * Cin + c];
  o[i] = val;
}
__global__ void ae_gru_init(float* __restrict__ h, unsigned short* __restrict__ hc, int n) {
  int i = blockIdx.x * blockDim.x + threadIdx.x;
  if (i < n) { h[i] = 0.f; hc[i] = 0; }
}
// One GRU step for both directions (dir0: t=j, dir1: t=511-j).
__global__ void ae_gru_gate(const float* __restrict__ gi, const float* __restrict__ gh,
                            const float* __restrict__ bhn, float* __restrict__ h,
                            unsigned short* __restrict__ hc, float* __restrict__ states,
                            unsigned short* __restrict__ states_bf, int j) {
  int i = blockIdx.x * blockDim.x + threadIdx.x;
  if (i >= 32768) return;                    // 2 dir * 32 batch * 512 feat
  int d = i >> 14, rem = i & 16383, b = rem >> 9, f = rem & 511;
  int t = d ? (511 - j) : j;
  size_t gir = ((size_t)b * 512 + t) * 1536;
  size_t ghr = (size_t)(d * 32 + b) * 1536;
  float r = sigm(gi[gir + f] + gh[ghr + f]);
  float z = sigm(gi[gir + 512 + f] + gh[ghr + 512 + f]);
  float n = tanhf(gi[gir + 1024 + f] + r * (gh[ghr + 1024 + f] + bhn[f]));
  size_t hi = (size_t)(d * 32 + b) * 512 + f;
  float hn = (1.f - z) * n + z * h[hi];
  h[hi] = hn;
  hc[hi] = f2b(hn);
  size_t si = ((size_t)b * 512 + t) * 1024 + (size_t)d * 512 + f;
  states[si] = hn;
  states_bf[si] = f2b(hn);
}
__global__ void ae_softmax(const float* __restrict__ s, unsigned short* __restrict__ p, int ncol) {
  int row = blockIdx.x, tid = threadIdx.x;
  const float* sr = s + (size_t)row * ncol;
  __shared__ float red[256];
  float m = -3.0e38f;
  for (int c = tid; c < ncol; c += 256) m = fmaxf(m, sr[c]);
  red[tid] = m; __syncthreads();
  for (int st = 128; st > 0; st >>= 1) {
    if (tid < st) red[tid] = fmaxf(red[tid], red[tid + st]);
    __syncthreads();
  }
  m = red[0]; __syncthreads();
  float sum = 0.f;
  for (int c = tid; c < ncol; c += 256) sum += expf(sr[c] - m);
  red[tid] = sum; __syncthreads();
  for (int st = 128; st > 0; st >>= 1) {
    if (tid < st) red[tid] += red[tid + st];
    __syncthreads();
  }
  float inv = 1.f / red[0];
  for (int c = tid; c < ncol; c += 256) p[(size_t)row * ncol + c] = f2b(expf(sr[c] - m) * inv);
}
__global__ void ae_layernorm(const float* __restrict__ y, const float* __restrict__ g,
                             const float* __restrict__ be, unsigned short* __restrict__ o, int C) {
  int row = blockIdx.x, tid = threadIdx.x;
  const float* yr = y + (size_t)row * C;
  __shared__ float r1[256], r2[256];
  float s1 = 0.f, s2 = 0.f;
  for (int c = tid; c < C; c += 256) { float v = yr[c]; s1 += v; s2 += v * v; }
  r1[tid] = s1; r2[tid] = s2; __syncthreads();
  for (int st = 128; st > 0; st >>= 1) {
    if (tid < st) { r1[tid] += r1[tid + st]; r2[tid] += r2[tid + st]; }
    __syncthreads();
  }
  float mu  = r1[0] / (float)C;
  float var = r2[0] / (float)C - mu * mu;
  float rs  = rsqrtf(var + 1e-6f);
  for (int c = tid; c < C; c += 256)
    o[(size_t)row * C + c] = f2b((yr[c] - mu) * rs * g[c] + be[c]);
}
__global__ void ae_finalnorm(float* __restrict__ o, int C) {
  int row = blockIdx.x, tid = threadIdx.x;   // blockDim == C == 256
  float v = o[(size_t)row * C + tid];
  __shared__ float red[256];
  red[tid] = v * v; __syncthreads();
  for (int st = 128; st > 0; st >>= 1) {
    if (tid < st) red[tid] += red[tid + st];
    __syncthreads();
  }
  float norm = sqrtf(red[0]);
  float sc = (norm > 1e-6f) ? 1.f / (norm + 1e-8f) : 1.f;
  o[(size_t)row * C + tid] = v * sc;
}

// ---------------------------------------------------------------------------
// Host side
// ---------------------------------------------------------------------------
static inline int cdiv(long long a, long long b) { return (int)((a + b - 1) / b); }

template<int MODE, bool ROWMAP>
static void gemmL(hipStream_t st, const unsigned short* A, const unsigned short* B,
                  float* outF, unsigned short* outB, const float* bias, const float* scale,
                  const float* res, int M, int N, int K, int lda, int ldb, int ldc,
                  float alpha = 0.f, int rowOuter = 0,
                  long long bsA = 0, long long bsB = 0, long long bsC = 0, int batches = 1) {
  const int nt4 = N >> 6;
  const int wpb = nt4 < 8 ? nt4 : 8;            // divides nt4 for all N used here
  dim3 grid((M >> 4) * (nt4 / wpb), batches, 1);
  ae_gemm<MODE, ROWMAP><<<grid, wpb * 32, 0, st>>>(
      A, B, outF, outB, bias, scale, res, M, N, K, lda, ldb, ldc,
      alpha, rowOuter, bsA, bsB, bsC);
}

extern "C" void kernel_launch(void* const* d_in, const int* in_sizes, int n_in,
                              void* d_out, int out_size, void* d_ws, size_t ws_size,
                              hipStream_t stream) {
  (void)in_sizes; (void)n_in; (void)out_size; (void)ws_size;
  using us = unsigned short;
  const float* IN[64];
  for (int i = 0; i < 51; ++i) IN[i] = (const float*)d_in[i];

  const int Bn = 32, S = 512;

  // ---- workspace arenas -------------------------------------------------
  char* base = (char*)d_ws;
  size_t off = 0;
  auto alloc = [&](size_t bytes) -> void* {
    void* p = base + off;
    off = (off + bytes + 255) & ~(size_t)255;
    return p;
  };
  us* w1T  = (us*)alloc((size_t)64 * 32 * 2);      // conv1 w, K padded 9->32
  us* w2T  = (us*)alloc((size_t)128 * 576 * 2);
  us* w3T  = (us*)alloc((size_t)256 * 1152 * 2);
  us* w4T  = (us*)alloc((size_t)512 * 2304 * 2);
  us* sk0T = (us*)alloc((size_t)512 * 64 * 2);
  us* sk1T = (us*)alloc((size_t)512 * 128 * 2);
  us* sk2T = (us*)alloc((size_t)512 * 256 * 2);
  us* wiT  = (us*)alloc((size_t)1536 * 512 * 2);
  us* whT  = (us*)alloc((size_t)1536 * 512 * 2);
  us* wqT  = (us*)alloc((size_t)1024 * 1024 * 2);
  us* wkT  = (us*)alloc((size_t)1024 * 1024 * 2);
  us* wvT  = (us*)alloc((size_t)1024 * 1024 * 2);
  us* woT  = (us*)alloc((size_t)1024 * 1024 * 2);
  us* wm1T = (us*)alloc((size_t)512 * 1024 * 2);
  us* wm2T = (us*)alloc((size_t)256 * 512 * 2);
  us* wm3T = (us*)alloc((size_t)256 * 256 * 2);
  float* bnsc[4]; float* bnsh[4];
  const int CC[4] = {64, 128, 256, 512};
  for (int i = 0; i < 4; ++i) { bnsc[i] = (float*)alloc(CC[i] * 4); bnsh[i] = (float*)alloc(CC[i] * 4); }
  us*   hin  = (us*)alloc((size_t)Bn * 8192 * 2);
  char* R1   = (char*)alloc((size_t)75497472);     // im2col arena; later q,k
  char* R2   = (char*)alloc((size_t)50331648);     // c1,c2,c3; later vT,scores,attn
  float* fused   = (float*)alloc((size_t)16384 * 512 * 4);  // later yln_bf
  us*    fusedbf = (us*)alloc((size_t)16384 * 512 * 2);     // later m1_bf
  char*  R3  = (char*)alloc((size_t)100663296);    // gi; later ctx_bf + y
  float* gh  = (float*)alloc((size_t)64 * 1536 * 4);
  float* hbuf = (float*)alloc((size_t)64 * 512 * 4);
  us*    hc   = (us*)alloc((size_t)64 * 512 * 2);
  float* states   = (float*)alloc((size_t)16384 * 1024 * 4);
  us*    statesbf = (us*)alloc((size_t)16384 * 1024 * 2);   // later m2_bf

  us* COL = (us*)R1;
  us* qbf = (us*)R1;
  us* kbf = (us*)(R1 + 33554432);
  us* c1 = (us*)R2;
  us* c2 = (us*)(R2 + 16777216);
  us* c3 = (us*)(R2 + 33554432);
  us* vT = (us*)R2;                                 // [B, 8, 128, 512] bf16
  float* scores = (float*)(R2 + 33554432);
  us* attnbf = (us*)(R2 + 41943040);
  float* gi = (float*)R3;
  us* ctxbf = (us*)R3;
  float* y = (float*)(R3 + 33554432);
  us* ylnbf = (us*)fused;
  us* m1bf  = (us*)fusedbf;
  us* m2bf  = (us*)statesbf;
  float* lat = (float*)d_out;

  auto EW = [&](int n) { return dim3(cdiv(n, 256)); };

  // ---- weight conversion (f32 -> bf16, transposed to [N,K]) -------------
  ae_zero_bf<<<EW(64 * 32), 256, 0, stream>>>(w1T, 64 * 32);
  ae_cast_t<<<EW(9 * 64), 256, 0, stream>>>(IN[1], w1T, 9, 64, 32);
  ae_cast_t<<<EW(576 * 128), 256, 0, stream>>>(IN[7], w2T, 576, 128, 576);
  ae_cast_t<<<EW(1152 * 256), 256, 0, stream>>>(IN[13], w3T, 1152, 256, 1152);
  ae_cast_t<<<EW(2304 * 512), 256, 0, stream>>>(IN[19], w4T, 2304, 512, 2304);
  ae_cast_t<<<EW(64 * 512), 256, 0, stream>>>(IN[25], sk0T, 64, 512, 64);
  ae_cast_t<<<EW(128 * 512), 256, 0, stream>>>(IN[27], sk1T, 128, 512, 128);
  ae_cast_t<<<EW(256 * 512), 256, 0, stream>>>(IN[29], sk2T, 256, 512, 256);
  ae_cast_t<<<EW(512 * 1536), 256, 0, stream>>>(IN[31], wiT, 512, 1536, 512);
  ae_cast_t<<<EW(512 * 1536), 256, 0, stream>>>(IN[33], whT, 512, 1536, 512);
  ae_cast_t<<<EW(1024 * 1024), 256, 0, stream>>>(IN[35], wqT, 1024, 1024, 1024);
  ae_cast_t<<<EW(1024 * 1024), 256, 0, stream>>>(IN[36], wkT, 1024, 1024, 1024);
  ae_cast_t<<<EW(1024 * 1024), 256, 0, stream>>>(IN[37], wvT, 1024, 1024, 1024);
  ae_cast_t<<<EW(1024 * 1024), 256, 0, stream>>>(IN[38], woT, 1024, 1024, 1024);
  ae_cast_t<<<EW(1024 * 512), 256, 0, stream>>>(IN[45], wm1T, 1024, 512, 1024);
  ae_cast_t<<<EW(512 * 256), 256, 0, stream>>>(IN[47], wm2T, 512, 256, 512);
  ae_cast_t<<<EW(256 * 256), 256, 0, stream>>>(IN[49], wm3T, 256, 256, 256);
  for (int i = 0; i < 4; ++i) {
    int bse = 1 + 6 * i;  // conv_w, conv_b, bn_s, bn_b, bn_m, bn_v
    ae_bnprep<<<EW(CC[i]), 256, 0, stream>>>(IN[bse + 1], IN[bse + 2], IN[bse + 3],
                                             IN[bse + 4], IN[bse + 5], bnsc[i], bnsh[i], CC[i]);
  }

  // ---- input scale + conv stack (im2col -> WMMA GEMM, fused BN+gelu) ----
  ae_prep_input<<<EW(Bn * 8192), 256, 0, stream>>>(IN[0], hin, Bn * 8192);

  ae_im2col<<<EW(Bn * 4096 * 32), 256, 0, stream>>>(hin, COL, Bn, 8192, 1, 4096, 32);
  gemmL<M_BN_GELU, false>(stream, COL, w1T, nullptr, c1, bnsh[0], bnsc[0], nullptr,
                          131072, 64, 32, 32, 32, 64);
  ae_im2col<<<EW(Bn * 2048 * 576), 256, 0, stream>>>(c1, COL, Bn, 4096, 64, 2048, 576);
  gemmL<M_BN_GELU, false>(stream, COL, w2T, nullptr, c2, bnsh[1], bnsc[1], nullptr,
                          65536, 128, 576, 576, 576, 128);
  ae_im2col<<<EW(Bn * 1024 * 1152), 256, 0, stream>>>(c2, COL, Bn, 2048, 128, 1024, 1152);
  gemmL<M_BN_GELU, false>(stream, COL, w3T, nullptr, c3, bnsh[2], bnsc[2], nullptr,
                          32768, 256, 1152, 1152, 1152, 256);
  ae_im2col<<<EW(Bn * 512 * 2304), 256, 0, stream>>>(c3, COL, Bn, 1024, 256, 512, 2304);
  gemmL<M_BN_GELU, false>(stream, COL, w4T, fused, nullptr, bnsh[3], bnsc[3], nullptr,
                          16384, 512, 2304, 2304, 2304, 512);

  // ---- skip fusion: fused += 0.1*(skips[i][:, :512, :] @ W + b) ---------
  gemmL<M_ACCUM, true>(stream, c1, sk0T, fused, nullptr, IN[26], nullptr, nullptr,
                       16384, 512, 64, 64, 64, 512, 0.1f, 4096);
  gemmL<M_ACCUM, true>(stream, c2, sk1T, fused, nullptr, IN[28], nullptr, nullptr,
                       16384, 512, 128, 128, 128, 512, 0.1f, 2048);
  gemmL<M_ACCUM, true>(stream, c3, sk2T, fused, nullptr, IN[30], nullptr, nullptr,
                       16384, 512, 256, 256, 256, 512, 0.1f, 1024);
  ae_cast<<<EW(16384 * 512), 256, 0, stream>>>(fused, fusedbf, 16384 * 512);

  // ---- GRU: precompute gi = fused @ Wi + bi, then 512 recurrent steps ---
  gemmL<M_BIAS, false>(stream, fusedbf, wiT, gi, nullptr, IN[32], nullptr, nullptr,
                       16384, 1536, 512, 512, 512, 1536);
  ae_gru_init<<<EW(32768), 256, 0, stream>>>(hbuf, hc, 32768);
  for (int j = 0; j < 512; ++j) {
    gemmL<M_PLAIN, false>(stream, hc, whT, gh, nullptr, nullptr, nullptr, nullptr,
                          64, 1536, 512, 512, 512, 1536);
    ae_gru_gate<<<EW(32768), 256, 0, stream>>>(gi, gh, IN[34], hbuf, hc, states, statesbf, j);
  }

  // ---- attention: QKV, per-batch per-head scores/softmax/ctx ------------
  gemmL<M_BIAS_BF16, false>(stream, statesbf, wqT, nullptr, qbf, IN[39], nullptr, nullptr,
                            16384, 1024, 1024, 1024, 1024, 1024);
  gemmL<M_BIAS_BF16, false>(stream, statesbf, wkT, nullptr, kbf, IN[40], nullptr, nullptr,
                            16384, 1024, 1024, 1024, 1024, 1024);
  gemmL<M_BIAS_BF16_VT, false>(stream, statesbf, wvT, nullptr, vT, IN[41], nullptr, nullptr,
                               16384, 1024, 1024, 1024, 1024, 1024);
  const float inv_sqrt_dh = 0.08838834764831845f;  // 1/sqrt(128)
  for (int b = 0; b < Bn; ++b) {
    const size_t boff = (size_t)b * S * 1024;
    // scores[h,q,k] = (q . k) / sqrt(Dh); K rows serve as B^T directly
    gemmL<M_ALPHA, false>(stream, qbf + boff, kbf + boff, scores, nullptr, nullptr,
                          nullptr, nullptr, 512, 512, 128, 1024, 1024, 512,
                          inv_sqrt_dh, 0, 128, 128, 262144, 8);
    ae_softmax<<<dim3(8 * 512), 256, 0, stream>>>(scores, attnbf, 512);
    // ctx[q, h*128+d] = attn @ v ; B^T = vT[b,h,:,:] (128 rows of 512)
    gemmL<M_BF16, false>(stream, attnbf, vT + (size_t)b * 8 * 128 * 512, nullptr,
                         ctxbf + boff, nullptr, nullptr, nullptr,
                         512, 128, 512, 512, 512, 1024,
                         0.f, 0, 262144, 65536, 128, 8);
  }
  // out projection + residual with states
  gemmL<M_BIAS_RES, false>(stream, ctxbf, woT, y, nullptr, IN[42], nullptr, states,
                           16384, 1024, 1024, 1024, 1024, 1024);

  // ---- layernorm + MLP + final unit-norm --------------------------------
  ae_layernorm<<<dim3(16384), 256, 0, stream>>>(y, IN[43], IN[44], ylnbf, 1024);
  gemmL<M_BIAS_GELU_BF16, false>(stream, ylnbf, wm1T, nullptr, m1bf, IN[46], nullptr,
                                 nullptr, 16384, 512, 1024, 1024, 1024, 512);
  gemmL<M_BIAS_GELU_BF16, false>(stream, m1bf, wm2T, nullptr, m2bf, IN[48], nullptr,
                                 nullptr, 16384, 256, 512, 512, 512, 256);
  gemmL<M_BIAS, false>(stream, m2bf, wm3T, lat, nullptr, IN[50], nullptr, nullptr,
                       16384, 256, 256, 256, 256, 256);
  ae_finalnorm<<<dim3(16384), 256, 0, stream>>>(lat, 256);
}
